// GenericNC_44633300140837
// MI455X (gfx1250) — matrix operand order repaired and checked by reading, not compiled, not run on hardware
//
#include <hip/hip_runtime.h>
#include <math.h>

#define N2      1024
#define BATCH   16
#define NSWEEPS 8

#define BM 128
#define BN 128
#define BK 32

typedef __attribute__((ext_vector_type(16))) __bf16 v16bf;
typedef __attribute__((ext_vector_type(8)))  __bf16 v8bf;
typedef __attribute__((ext_vector_type(8)))  float  v8f;

union FragAB { v16bf v; v8bf h[2]; };

__device__ __forceinline__ unsigned short f32_to_bf16(float x) {
  unsigned u = __float_as_uint(x);
  u += 0x7FFFu + ((u >> 16) & 1u);        // round-to-nearest-even
  return (unsigned short)(u >> 16);
}
__device__ __forceinline__ float bf16_to_f32(unsigned short h) {
  return __uint_as_float(((unsigned)h) << 16);
}
// split x into bf16 hi + bf16 lo (bf16x3 fp32-emulation operands)
__device__ __forceinline__ void split_bf16(float x, unsigned short& hi, unsigned short& lo) {
  const unsigned short h = f32_to_bf16(x);
  hi = h;
  lo = f32_to_bf16(x - bf16_to_f32(h));
}
__device__ __forceinline__ unsigned pack2(unsigned short a, unsigned short b) {
  return (unsigned)a | ((unsigned)b << 16);
}

// ---------------------------------------------------------------------------
// Kernel 1: batched SYRK  S[b] = Z[b] * Z[b]^T  via bf16x3-split WMMA.
// Block = 256 threads (8 waves); each block computes a 128x128 tile of S.
//   stage 1: async DMA (global_load_async_to_lds_b128, ASYNCcnt) fills
//            double-buffered raw f32 tiles in LDS (no VGPR staging).
//   stage 2: one conversion pass splits each f32 element exactly once into
//            shared bf16 hi/lo tiles (packed uint2 stores).
//   stage 3: WMMA fragments = two contiguous 16B ds_load_b128 per fragment.
// ---------------------------------------------------------------------------
__global__ __launch_bounds__(256)
void syrk_bf16x3_wmma(const float* __restrict__ Z, float* __restrict__ S) {
  const int batch = blockIdx.z;
  const int i0 = blockIdx.y * BM;
  const int j0 = blockIdx.x * BN;
  const float* Zb = Z + (size_t)batch * N2 * N2;
  float*       Sb = S + (size_t)batch * N2 * N2;

  __shared__ __align__(16) float Af32[2][BM][BK];       // async-DMA staging
  __shared__ __align__(16) float Bf32[2][BN][BK];
  __shared__ __align__(16) unsigned short Ah[BM][BK], Al[BM][BK];  // bf16 tiles
  __shared__ __align__(16) unsigned short Bh[BN][BK], Bl[BN][BK];

  const int tid   = threadIdx.x;
  const int wave  = tid >> 5;
  const int lane  = tid & 31;
  const int wm    = wave & 3;      // 4 strips of 32 rows
  const int wn    = wave >> 2;     // 2 strips of 64 cols
  const int lhalf = lane >> 4;     // lane half: 0 / 1
  const int l16   = lane & 15;

  // 8 async b128 DMAs per thread per tile pair (4 for A, 4 for B).
  auto issue_tile = [&](int buf, int kko) {
    for (int e = tid; e < (BM * BK) / 4; e += 256) {
      const int idx = e * 4;
      const int row = idx >> 5;       // BK == 32
      const int col = idx & 31;
      const unsigned la = (unsigned)(size_t)&Af32[buf][row][col];
      const unsigned long long ga =
          (unsigned long long)(size_t)(Zb + (size_t)(i0 + row) * N2 + kko + col);
      asm volatile("global_load_async_to_lds_b128 %0, %1, off"
                   :: "v"(la), "v"(ga) : "memory");
      const unsigned lb = (unsigned)(size_t)&Bf32[buf][row][col];
      const unsigned long long gb =
          (unsigned long long)(size_t)(Zb + (size_t)(j0 + row) * N2 + kko + col);
      asm volatile("global_load_async_to_lds_b128 %0, %1, off"
                   :: "v"(lb), "v"(gb) : "memory");
    }
  };

  v8f acc[2][4] = {};

  issue_tile(0, 0);                          // prologue: fill buffer 0

  for (int kk = 0; kk < N2; kk += BK) {
    const int cur = (kk >> 5) & 1;

    if (kk + BK < N2) {
      issue_tile(cur ^ 1, kk + BK);          // prefetch next tile
      // async loads retire in order: threshold-8 == "older (current) group done"
      asm volatile("s_wait_asynccnt 8" ::: "memory");
    } else {
      asm volatile("s_wait_asynccnt 0" ::: "memory");
    }
    __syncthreads();   // staging[cur] visible; prior bf16-tile reads all done

    // Conversion pass: each f32 element split exactly once, packed stores.
    for (int e = tid; e < (BM * BK) / 4; e += 256) {
      const int idx = e * 4;
      const int row = idx >> 5;
      const int col = idx & 31;
      const float4 a = *(const float4*)&Af32[cur][row][col];
      const float4 b = *(const float4*)&Bf32[cur][row][col];
      unsigned short h0, h1, h2, h3, l0, l1, l2, l3;
      split_bf16(a.x, h0, l0); split_bf16(a.y, h1, l1);
      split_bf16(a.z, h2, l2); split_bf16(a.w, h3, l3);
      *(uint2*)&Ah[row][col] = make_uint2(pack2(h0, h1), pack2(h2, h3));
      *(uint2*)&Al[row][col] = make_uint2(pack2(l0, l1), pack2(l2, l3));
      split_bf16(b.x, h0, l0); split_bf16(b.y, h1, l1);
      split_bf16(b.z, h2, l2); split_bf16(b.w, h3, l3);
      *(uint2*)&Bh[row][col] = make_uint2(pack2(h0, h1), pack2(h2, h3));
      *(uint2*)&Bl[row][col] = make_uint2(pack2(l0, l1), pack2(l2, l3));
    }
    __syncthreads();   // bf16 tiles ready; staging[cur] reads complete

    // A fragments (16-bit A 16x32 layout: lanes 0-15 K=0..7,16..23;
    // lanes 16-31 K=8..15,24..31 -> two 16B chunks at [M][kb], [M][kb+16]).
    FragAB a_hi[2], a_lo[2];
#pragma unroll
    for (int mt = 0; mt < 2; ++mt) {
      const int M  = wm * 32 + mt * 16 + l16;
      const int kb = lhalf * 8;
      a_hi[mt].h[0] = *(const v8bf*)&Ah[M][kb];
      a_hi[mt].h[1] = *(const v8bf*)&Ah[M][kb + 16];
      a_lo[mt].h[0] = *(const v8bf*)&Al[M][kb];
      a_lo[mt].h[1] = *(const v8bf*)&Al[M][kb + 16];
    }

#pragma unroll
    for (int nt = 0; nt < 4; ++nt) {
      // B fragment (16-bit B 32x16: lanes 0-15 K=0..7,16..23, lanes 16-31 K=8..15,24..31,
      // N = lane&15) -> two 16B chunks at [N][lhalf*8], [N][lhalf*8+16].
      const int Ncol = wn * 64 + nt * 16 + l16;
      const int kb   = lhalf * 8;
      FragAB b_hi, b_lo;
      b_hi.h[0] = *(const v8bf*)&Bh[Ncol][kb];
      b_hi.h[1] = *(const v8bf*)&Bh[Ncol][kb + 16];
      b_lo.h[0] = *(const v8bf*)&Bl[Ncol][kb];
      b_lo.h[1] = *(const v8bf*)&Bl[Ncol][kb + 16];
#pragma unroll
      for (int mt = 0; mt < 2; ++mt) {
        v8f c = acc[mt][nt];
        c = __builtin_amdgcn_wmma_f32_16x16x32_bf16(false, a_hi[mt].v, false, b_hi.v, (short)0, c, false, false);
        c = __builtin_amdgcn_wmma_f32_16x16x32_bf16(false, a_hi[mt].v, false, b_lo.v, (short)0, c, false, false);
        c = __builtin_amdgcn_wmma_f32_16x16x32_bf16(false, a_lo[mt].v, false, b_hi.v, (short)0, c, false, false);
        acc[mt][nt] = c;
      }
    }
  }

  // Store C tiles: 32-bit C/D layout: VGPR r -> row (lhalf*8 + r), col = lane&15.
#pragma unroll
  for (int mt = 0; mt < 2; ++mt)
#pragma unroll
    for (int nt = 0; nt < 4; ++nt) {
      const int col   = j0 + wn * 64 + nt * 16 + l16;
      const int rbase = i0 + wm * 32 + mt * 16 + lhalf * 8;
#pragma unroll
      for (int r = 0; r < 8; ++r)
        Sb[(size_t)(rbase + r) * N2 + col] = acc[mt][nt][r];
    }
}

// ---------------------------------------------------------------------------
// Kernel 2: V = I per batch.
// ---------------------------------------------------------------------------
__global__ void init_identity(float* __restrict__ V) {
  const size_t i      = (size_t)blockIdx.x * blockDim.x + threadIdx.x;
  const size_t within = i & (size_t)(N2 * N2 - 1);
  const int row = (int)(within >> 10);
  const int col = (int)(within & (N2 - 1));
  V[i] = (row == col) ? 1.0f : 0.0f;
}

// ---------------------------------------------------------------------------
// Kernel 3: persistent two-sided cyclic Jacobi, one workgroup per matrix.
// Round-robin tournament scheduling: 512 disjoint (p,q) pairs per round,
// N2-1 rounds per sweep. A lives in d_ws, V in d_out; both fit in L2.
// ---------------------------------------------------------------------------
__global__ __launch_bounds__(1024)
void jacobi_sweeps(float* Aall, float* Vall) {
  const int b = blockIdx.x;
  float* A = Aall + (size_t)b * N2 * N2;
  float* V = Vall + (size_t)b * N2 * N2;

  __shared__ float cs[N2 / 2], sn[N2 / 2];
  __shared__ short pp[N2 / 2], qq[N2 / 2];

  const int tid = threadIdx.x;

  for (int sweep = 0; sweep < NSWEEPS; ++sweep) {
    for (int r = 0; r < N2 - 1; ++r) {
      // Phase 0: pairing + rotation angles (512 threads).
      if (tid < N2 / 2) {
        int p, q;
        if (tid == 0) {
          p = N2 - 1;
          q = r % (N2 - 1);
        } else {
          p = (r + tid) % (N2 - 1);
          q = (r + (N2 - 1) - tid) % (N2 - 1);
        }
        if (p > q) { const int t = p; p = q; q = t; }
        pp[tid] = (short)p;
        qq[tid] = (short)q;

        const float app = A[(size_t)p * N2 + p];
        const float aqq = A[(size_t)q * N2 + q];
        const float apq = A[(size_t)p * N2 + q];
        float c = 1.0f, s = 0.0f;
        if (fabsf(apq) > 1e-30f) {
          const float theta = 0.5f * (aqq - app) / apq;
          const float t = copysignf(1.0f, theta) /
                          (fabsf(theta) + sqrtf(1.0f + theta * theta));
          c = 1.0f / sqrtf(1.0f + t * t);
          s = t * c;
        }
        cs[tid] = c;
        sn[tid] = s;
      }
      __syncthreads();

      // Phase 1: row update  A <- G^T A. (pair,col) work items are disjoint.
      for (int idx = tid; idx < (N2 / 2) * N2; idx += 1024) {
        const int k   = idx >> 10;
        const int col = idx & (N2 - 1);
        const int p = pp[k], q = qq[k];
        const float c = cs[k], s = sn[k];
        const float ap = A[(size_t)p * N2 + col];
        const float aq = A[(size_t)q * N2 + col];
        A[(size_t)p * N2 + col] = c * ap - s * aq;
        A[(size_t)q * N2 + col] = s * ap + c * aq;
      }
      __syncthreads();

      // Phase 2: column update  A <- A G  and accumulate  V <- V G.
      for (int idx = tid; idx < (N2 / 2) * N2; idx += 1024) {
        const int k   = idx >> 10;
        const int row = idx & (N2 - 1);
        const int p = pp[k], q = qq[k];
        const float c = cs[k], s = sn[k];
        const float ap = A[(size_t)row * N2 + p];
        const float aq = A[(size_t)row * N2 + q];
        A[(size_t)row * N2 + p] = c * ap - s * aq;
        A[(size_t)row * N2 + q] = s * ap + c * aq;
        const float vp = V[(size_t)row * N2 + p];
        const float vq = V[(size_t)row * N2 + q];
        V[(size_t)row * N2 + p] = c * vp - s * vq;
        V[(size_t)row * N2 + q] = s * vp + c * vq;
      }
      __syncthreads();
    }
  }
}

// ---------------------------------------------------------------------------
// Kernel 4: rank eigenvalues (diag of A) ascending, permute V's columns into
// the (now dead) A workspace. Out aliases A on purpose: the only reads of A
// (the diagonal) happen before the barrier, into LDS.
// ---------------------------------------------------------------------------
__global__ __launch_bounds__(1024)
void sort_permute(const float* Adiag, const float* __restrict__ V, float* Out) {
  const int b = blockIdx.x;
  const float* Ab = Adiag + (size_t)b * N2 * N2;
  const float* Vb = V     + (size_t)b * N2 * N2;
  float*       Ob = Out   + (size_t)b * N2 * N2;

  __shared__ float d[N2];
  const int j = threadIdx.x;
  d[j] = Ab[(size_t)j * N2 + j];
  __syncthreads();

  const float dj = d[j];
  int rank = 0;
  for (int i = 0; i < N2; ++i) {
    const float di = d[i];
    rank += (di < dj) || (di == dj && i < j);
  }

  for (int row = 0; row < N2; ++row)
    Ob[(size_t)row * N2 + rank] = Vb[(size_t)row * N2 + j];
}

// ---------------------------------------------------------------------------
extern "C" void kernel_launch(void* const* d_in, const int* in_sizes, int n_in,
                              void* d_out, int out_size, void* d_ws, size_t ws_size,
                              hipStream_t stream) {
  const float* z = (const float*)d_in[0];        // (16, 1024*1024) f32
  float* out = (float*)d_out;                    // (16, 1024, 1024) f32 eigenvectors
  float* A   = (float*)d_ws;                     // 16 * 1M f32 = 64 MB workspace

  // 1) S = Z Z^T (symmetric by construction; 0.5*(S+S^T) is the identity here)
  dim3 gGemm(N2 / BN, N2 / BM, BATCH);
  syrk_bf16x3_wmma<<<gGemm, 256, 0, stream>>>(z, A);

  // 2) V = I
  init_identity<<<(BATCH * N2 * N2) / 256, 256, 0, stream>>>(out);

  // 3) Jacobi eigendecomposition: A -> diag(eigvals), V -> eigenvectors
  jacobi_sweeps<<<BATCH, 1024, 0, stream>>>(A, out);

  // 4) Sort columns by ascending eigenvalue into the A region, copy back.
  sort_permute<<<BATCH, N2, 0, stream>>>(A, out, A);
  hipMemcpyAsync(d_out, d_ws, (size_t)BATCH * N2 * N2 * sizeof(float),
                 hipMemcpyDeviceToDevice, stream);
}
